// NMTJointDecoder_63582695850630
// MI455X (gfx1250) — matrix operand order repaired and checked by reading, compiled, not run on hardware
//
#include <hip/hip_runtime.h>
#include <hip/hip_bf16.h>
#include <math.h>

// ---------------------------------------------------------------------------
// NMT joint decoder (slot+intent) for gfx1250.
// B=64 T=512 H=768 E=256 SLOT=256 INTENT=128, XK = E+2H = 1792.
// GEMMs: bf16 WMMA (v_wmma_f32_16x16x32_bf16), f32 accumulate.
// Tile staging: global_load_async_to_lds_b128 (ASYNCcnt) instead of
// load->wait->ds_store bounce through VGPRs.
// ---------------------------------------------------------------------------

#define Bn      64
#define Tn      512
#define Hn      768
#define En      256
#define SLOTn   256
#define INTENTn 128
#define XKn     1792   // E + 2H
#define G4n     3072   // 4H

typedef __attribute__((ext_vector_type(16))) __bf16 v16bf;
typedef __attribute__((ext_vector_type(8)))  __bf16 v8bf;
typedef __attribute__((ext_vector_type(8)))  float  v8f;

union AFragU { v16bf v; v8bf h[2]; };

// A fragment (16x32 bf16, contiguous k = 16*(lane>>4) + s, matching B): rowPtr points at
// (row m = mt*16 + (lane&15), column kb) of a row-major [M][K] bf16 matrix.
__device__ __forceinline__ v16bf load_a_row(const __bf16* rowPtr, int lane) {
    const int half = (lane >> 4) * 16;  // 0 or 16
    AFragU u;
    u.h[0] = *(const v8bf*)(rowPtr + half);
    u.h[1] = *(const v8bf*)(rowPtr + half + 8);
    return u.v;
}

// B fragment (32x16 bf16): W is row-major [N][K]; B[k][n] = W[n][k].
__device__ __forceinline__ v16bf load_b_frag(const __bf16* Wn0, int ldk, int kb, int lane) {
    const int n    = lane & 15;
    const int koff = kb + ((lane >> 4) << 4);  // +0 or +16
    return *(const v16bf*)(Wn0 + (size_t)n * ldk + koff);
}

__device__ __forceinline__ v8f wmma_bf16(v16bf a, v16bf b, v8f c) {
    return __builtin_amdgcn_wmma_f32_16x16x32_bf16(false, a, false, b, (short)0, c, false, false);
}

// Async DMA: 16B global -> LDS per active lane, tracked by ASYNCcnt.
__device__ __forceinline__ void async_copy_b128(const __bf16* gsrc, __bf16* ldst) {
    unsigned lds = (unsigned)(unsigned long long)(void*)ldst;          // low 32b = LDS offset
    unsigned long long ga = (unsigned long long)(const void*)gsrc;
    asm volatile("global_load_async_to_lds_b128 %0, %1, off"
                 :: "v"(lds), "v"(ga) : "memory");
}
__device__ __forceinline__ void wait_async0() {
    asm volatile("s_wait_asynccnt 0x0" ::: "memory");
}

__device__ __forceinline__ float sigmoidf_(float x) { return 1.0f / (1.0f + __expf(-x)); }

// ---------------------------------------------------------------------------
// f32 -> bf16 conversion
// ---------------------------------------------------------------------------
__global__ void cvt_f32_to_bf16(const float* __restrict__ src, __bf16* __restrict__ dst, int n) {
    int i = blockIdx.x * blockDim.x + threadIdx.x;
    if (i < n) dst[i] = (__bf16)src[i];
}

// ---------------------------------------------------------------------------
// energies[b,t,:] = enc[b,t,:] @ W_attn^T + b_attn     (M=32768, N=768, K=768)
// block = 128 threads (4 waves); block computes a 64(M) x 64(N) tile.
// ---------------------------------------------------------------------------
__global__ __launch_bounds__(128) void energies_kernel(
    const __bf16* __restrict__ encb, const __bf16* __restrict__ Wat,
    const float* __restrict__ b_attn, __bf16* __restrict__ energ) {
    __shared__ __bf16 at[64 * 32];
    const int tid = threadIdx.x, lane = tid & 31, wv = tid >> 5;
    const int m0 = blockIdx.x * 64;
    const int n0 = blockIdx.y * 64 + wv * 16;
    const __bf16* Wn = Wat + (size_t)n0 * Hn;
    v8f acc[4] = {v8f{0.f,0.f,0.f,0.f,0.f,0.f,0.f,0.f}, v8f{0.f,0.f,0.f,0.f,0.f,0.f,0.f,0.f},
                  v8f{0.f,0.f,0.f,0.f,0.f,0.f,0.f,0.f}, v8f{0.f,0.f,0.f,0.f,0.f,0.f,0.f,0.f}};
    for (int kb = 0; kb < Hn; kb += 32) {
        // stage 64x32 A tile: 256 16B chunks, 2 per thread, async DMA to LDS
        #pragma unroll
        for (int c2 = 0; c2 < 2; c2++) {
            int c = tid + c2 * 128;
            int r = c >> 2, kc = (c & 3) * 8;
            async_copy_b128(encb + ((size_t)(m0 + r)) * Hn + kb + kc, &at[r * 32 + kc]);
        }
        wait_async0();
        __syncthreads();
        v16bf bf = load_b_frag(Wn, Hn, kb, lane);
        #pragma unroll
        for (int mt = 0; mt < 4; mt++) {
            v16bf af = load_a_row(&at[(mt * 16 + (lane & 15)) * 32], lane);
            acc[mt] = wmma_bf16(af, bf, acc[mt]);
        }
        __syncthreads();
    }
    const int n = lane & 15;
    const float bias = b_attn[n0 + n];
    #pragma unroll
    for (int mt = 0; mt < 4; mt++)
        #pragma unroll
        for (int v = 0; v < 8; v++) {
            int m = m0 + mt * 16 + v + 8 * (lane >> 4);
            energ[(size_t)m * Hn + n0 + n] = (__bf16)(acc[mt][v] + bias);
        }
}

// ---------------------------------------------------------------------------
// init: lengths, ctx0 = enc[b, len-1], pack x0 = [0 | ctx0 | enc[:,0,:]] bf16
// ---------------------------------------------------------------------------
__global__ __launch_bounds__(256) void init_kernel(
    const float* __restrict__ enc, const int* __restrict__ mask, __bf16* __restrict__ x0) {
    const int b = blockIdx.x, tid = threadIdx.x;
    __shared__ int red[256];
    int s = 0;
    for (int t = tid; t < Tn; t += 256) s += mask[b * Tn + t];
    red[tid] = s; __syncthreads();
    for (int w = 128; w > 0; w >>= 1) { if (tid < w) red[tid] += red[tid + w]; __syncthreads(); }
    const int len  = red[0];
    const int last = (len > 0) ? (len - 1) : 0;
    const float* c0 = enc + ((size_t)b * Tn + last) * Hn;
    const float* a0 = enc + (size_t)b * Tn * Hn;     // aligned for step 0
    __bf16* xr = x0 + (size_t)b * XKn;
    for (int k = tid; k < En; k += 256) xr[k] = (__bf16)0.0f;
    for (int k = tid; k < Hn; k += 256) {
        xr[En + k]      = (__bf16)c0[k];
        xr[En + Hn + k] = (__bf16)a0[k];
    }
}

// ---------------------------------------------------------------------------
// LSTM step: g = x @ W_ih^T + b; h = sigmoid(o)*tanh(sigmoid(i)*tanh(g)).
// f-gate skipped (c_prev = 0).  block = 128 threads = 4 waves: waves 0-2 own
// gates i,g,o (full-EXEC WMMA), wave 3 helps stage tiles.  grid = H/16 = 48.
// ---------------------------------------------------------------------------
__global__ __launch_bounds__(128) void lstm_kernel(
    const __bf16* __restrict__ x, const __bf16* __restrict__ Wih,
    const float* __restrict__ b_ih, const float* __restrict__ b_hh,
    float* __restrict__ hf, __bf16* __restrict__ hb) {
    __shared__ __bf16 xt[64 * 32];
    __shared__ float  act[3][64 * 16];
    const int tid = threadIdx.x, lane = tid & 31, wv = tid >> 5;
    const int n0 = blockIdx.x * 16;
    const int gateOff = (wv == 1) ? 2 * Hn : ((wv == 2) ? 3 * Hn : 0);  // i, g, o rows
    const __bf16* Wn = Wih + (size_t)(gateOff + n0) * XKn;
    v8f acc[4] = {v8f{0.f,0.f,0.f,0.f,0.f,0.f,0.f,0.f}, v8f{0.f,0.f,0.f,0.f,0.f,0.f,0.f,0.f},
                  v8f{0.f,0.f,0.f,0.f,0.f,0.f,0.f,0.f}, v8f{0.f,0.f,0.f,0.f,0.f,0.f,0.f,0.f}};
    for (int kb = 0; kb < XKn; kb += 32) {
        // stage 64x32 x tile: 256 16B chunks, 2 per thread, async DMA to LDS
        #pragma unroll
        for (int c2 = 0; c2 < 2; c2++) {
            int c = tid + c2 * 128;
            int r = c >> 2, kc = (c & 3) * 8;
            async_copy_b128(x + (size_t)r * XKn + kb + kc, &xt[r * 32 + kc]);
        }
        wait_async0();
        __syncthreads();
        if (wv < 3) {
            if (kb + 32 < XKn)                           // pull next W_ih k-slice
                __builtin_prefetch(Wn + (size_t)(lane & 15) * XKn + kb + 32, 0, 0);
            v16bf bf = load_b_frag(Wn, XKn, kb, lane);
            #pragma unroll
            for (int mt = 0; mt < 4; mt++) {
                v16bf af = load_a_row(&xt[(mt * 16 + (lane & 15)) * 32], lane);
                acc[mt] = wmma_bf16(af, bf, acc[mt]);
            }
        }
        __syncthreads();
    }
    if (wv < 3) {
        const int n = lane & 15;
        const int grow = gateOff + n0 + n;
        const float bias = b_ih[grow] + b_hh[grow];
        #pragma unroll
        for (int mt = 0; mt < 4; mt++)
            #pragma unroll
            for (int v = 0; v < 8; v++) {
                int m = mt * 16 + v + 8 * (lane >> 4);
                float g = acc[mt][v] + bias;
                act[wv][m * 16 + n] = (wv == 1) ? tanhf(g) : sigmoidf_(g);
            }
    }
    __syncthreads();
    for (int i = tid; i < 1024; i += 128) {
        int m = i >> 4, nn = i & 15;
        float c = act[0][i] * act[1][i];
        float h = act[2][i] * tanhf(c);
        hf[(size_t)m * Hn + n0 + nn] = h;
        hb[(size_t)m * Hn + n0 + nn] = (__bf16)h;
    }
}

// ---------------------------------------------------------------------------
// slot scores: [h | ctx] @ W_slot^T + b_slot   (M=64, N=256, K=1536)
// ctx is read from x_cur columns [E, E+H).  1 wave/block, 16 blocks.
// ---------------------------------------------------------------------------
__global__ __launch_bounds__(32) void slot_gemm(
    const __bf16* __restrict__ hb, const __bf16* __restrict__ xcur,
    const __bf16* __restrict__ Wslot, const float* __restrict__ b_slot,
    float* __restrict__ score) {
    const int lane = threadIdx.x & 31;
    const int n0 = blockIdx.x * 16;
    const __bf16* Wn = Wslot + (size_t)n0 * (2 * Hn);
    v8f acc[4] = {v8f{0.f,0.f,0.f,0.f,0.f,0.f,0.f,0.f}, v8f{0.f,0.f,0.f,0.f,0.f,0.f,0.f,0.f},
                  v8f{0.f,0.f,0.f,0.f,0.f,0.f,0.f,0.f}, v8f{0.f,0.f,0.f,0.f,0.f,0.f,0.f,0.f}};
    for (int kb = 0; kb < 2 * Hn; kb += 32) {
        v16bf bf = load_b_frag(Wn, 2 * Hn, kb, lane);
        #pragma unroll
        for (int mt = 0; mt < 4; mt++) {
            int m = mt * 16 + (lane & 15);
            const __bf16* row = (kb < Hn)
                ? (hb   + (size_t)m * Hn  + kb)
                : (xcur + (size_t)m * XKn + En + (kb - Hn));
            acc[mt] = wmma_bf16(load_a_row(row, lane), bf, acc[mt]);
        }
    }
    const int n = lane & 15;
    const float bias = b_slot[n0 + n];
    #pragma unroll
    for (int mt = 0; mt < 4; mt++)
        #pragma unroll
        for (int v = 0; v < 8; v++) {
            int m = mt * 16 + v + 8 * (lane >> 4);
            score[(size_t)m * SLOTn + n0 + n] = acc[mt][v] + bias;
        }
}

// ---------------------------------------------------------------------------
// log_softmax + greedy argmax + embed feedback + next aligned slice.
// one block per batch row.
// ---------------------------------------------------------------------------
__global__ __launch_bounds__(256) void slot_finish(
    const float* __restrict__ score, const float* __restrict__ emb,
    const __bf16* __restrict__ encb, float* __restrict__ out,
    __bf16* __restrict__ xnext, int t) {
    const int b = blockIdx.x, tid = threadIdx.x;
    __shared__ float rv[256];
    __shared__ int   ri[256];
    const float s = score[(size_t)b * SLOTn + tid];
    rv[tid] = s; ri[tid] = tid; __syncthreads();
    for (int w = 128; w > 0; w >>= 1) {                   // argmax, first-index ties
        if (tid < w) {
            float a = rv[tid], c = rv[tid + w];
            if (c > a || (c == a && ri[tid + w] < ri[tid])) { rv[tid] = c; ri[tid] = ri[tid + w]; }
        }
        __syncthreads();
    }
    const float mx = rv[0];
    const int amax = ri[0];
    __syncthreads();
    rv[tid] = __expf(s - mx); __syncthreads();
    for (int w = 128; w > 0; w >>= 1) { if (tid < w) rv[tid] += rv[tid + w]; __syncthreads(); }
    const float lse = logf(rv[0]);
    out[(((size_t)b * Tn) + t) * SLOTn + tid] = s - mx - lse;
    // greedy embedding feedback into next x
    xnext[(size_t)b * XKn + tid] = (__bf16)emb[(size_t)amax * En + tid];
    // aligned input for step t+1
    if (t + 1 < Tn)
        for (int k = tid; k < Hn; k += 256)
            xnext[(size_t)b * XKn + En + Hn + k] = encb[((size_t)b * Tn + t + 1) * Hn + k];
}

// ---------------------------------------------------------------------------
// attention(h): e = energies . h ; masked softmax over T ; ctx = alpha @ enc
// one block per batch row.
// ---------------------------------------------------------------------------
__global__ __launch_bounds__(256) void attn_kernel(
    const __bf16* __restrict__ energ, const __bf16* __restrict__ hb,
    const float* __restrict__ enc, const int* __restrict__ mask,
    float* __restrict__ ctx, __bf16* __restrict__ xnext) {
    const int b = blockIdx.x, tid = threadIdx.x;
    __shared__ float hv[Hn];
    __shared__ float e[Tn];
    __shared__ float red[256];
    for (int k = tid; k < Hn; k += 256) hv[k] = (float)hb[(size_t)b * Hn + k];
    __syncthreads();
    for (int s = tid; s < Tn; s += 256) {
        const __bf16* row = energ + ((size_t)b * Tn + s) * Hn;
        float a = 0.f;
        for (int k = 0; k < Hn; k += 8) {
            v8bf v = *(const v8bf*)(row + k);
            #pragma unroll
            for (int j = 0; j < 8; j++) a += (float)v[j] * hv[k + j];
        }
        e[s] = (mask[b * Tn + s] == 0) ? -1e12f : a;
    }
    __syncthreads();
    red[tid] = fmaxf(e[tid], e[tid + 256]); __syncthreads();
    for (int w = 128; w > 0; w >>= 1) { if (tid < w) red[tid] = fmaxf(red[tid], red[tid + w]); __syncthreads(); }
    const float mx = red[0]; __syncthreads();
    float s0 = __expf(e[tid] - mx), s1 = __expf(e[tid + 256] - mx);
    e[tid] = s0; e[tid + 256] = s1;
    red[tid] = s0 + s1; __syncthreads();
    for (int w = 128; w > 0; w >>= 1) { if (tid < w) red[tid] += red[tid + w]; __syncthreads(); }
    const float inv = 1.f / red[0]; __syncthreads();
    e[tid] *= inv; e[tid + 256] *= inv;
    __syncthreads();
    for (int k = tid; k < Hn; k += 256) {
        float a = 0.f;
        const float* col = enc + (size_t)b * Tn * Hn + k;
        for (int s = 0; s < Tn; s++) a += e[s] * col[(size_t)s * Hn];
        ctx[(size_t)b * Hn + k] = a;
        xnext[(size_t)b * XKn + En + k] = (__bf16)a;
    }
}

__global__ void save_h0(const float* __restrict__ hf, const float* __restrict__ ctxf,
                        float* __restrict__ h0, float* __restrict__ c1) {
    int i = blockIdx.x * 256 + threadIdx.x;
    if (i < Bn * Hn) { h0[i] = hf[i]; c1[i] = ctxf[i]; }
}

// ---------------------------------------------------------------------------
// intent: relu([h0|ctx1] @ W_i1^T + b_i1) @ W_i2^T + b_i2  (tiny; f32 VALU)
// ---------------------------------------------------------------------------
__global__ __launch_bounds__(128) void intent_kernel(
    const float* __restrict__ h0, const float* __restrict__ c1,
    const float* __restrict__ W1, const float* __restrict__ b1,
    const float* __restrict__ W2, const float* __restrict__ b2,
    float* __restrict__ out) {
    const int b = blockIdx.x, tid = threadIdx.x;
    __shared__ float in[2 * Hn];
    __shared__ float hid[Hn / 2];
    for (int k = tid; k < Hn; k += 128) {
        in[k]      = h0[(size_t)b * Hn + k];
        in[Hn + k] = c1[(size_t)b * Hn + k];
    }
    __syncthreads();
    for (int j = tid; j < Hn / 2; j += 128) {
        const float* w = W1 + (size_t)j * (2 * Hn);
        float a = b1[j];
        for (int k = 0; k < 2 * Hn; k++) a += w[k] * in[k];
        hid[j] = fmaxf(a, 0.f);
    }
    __syncthreads();
    if (tid < INTENTn) {
        const float* w = W2 + (size_t)tid * (Hn / 2);
        float a = b2[tid];
        for (int k = 0; k < Hn / 2; k++) a += w[k] * hid[k];
        out[(size_t)Bn * Tn * SLOTn + (size_t)b * INTENTn + tid] = a;
    }
}

// ---------------------------------------------------------------------------
extern "C" void kernel_launch(void* const* d_in, const int* in_sizes, int n_in,
                              void* d_out, int out_size, void* d_ws, size_t ws_size,
                              hipStream_t stream) {
    (void)in_sizes; (void)n_in; (void)out_size; (void)ws_size;
    const float* enc    = (const float*)d_in[0];
    const int*   mask   = (const int*)d_in[1];
    const float* emb    = (const float*)d_in[2];
    const float* W_ih   = (const float*)d_in[3];
    const float* b_ih   = (const float*)d_in[4];
    const float* b_hh   = (const float*)d_in[5];
    const float* W_attn = (const float*)d_in[6];
    const float* b_attn = (const float*)d_in[7];
    const float* W_slot = (const float*)d_in[8];
    const float* b_slot = (const float*)d_in[9];
    const float* W_i1   = (const float*)d_in[10];
    const float* b_i1   = (const float*)d_in[11];
    const float* W_i2   = (const float*)d_in[12];
    const float* b_i2   = (const float*)d_in[13];
    float* out = (float*)d_out;

    char* ws = (char*)d_ws;
    auto carve = [&](size_t bytes) { char* p = ws; ws += (bytes + 255) & ~(size_t)255; return p; };
    __bf16* enc_b   = (__bf16*)carve((size_t)Bn * Tn * Hn * 2);      // 50.3 MB
    __bf16* energ_b = (__bf16*)carve((size_t)Bn * Tn * Hn * 2);      // 50.3 MB
    __bf16* Wih_b   = (__bf16*)carve((size_t)G4n * XKn * 2);         // 11.0 MB
    __bf16* Wsl_b   = (__bf16*)carve((size_t)SLOTn * 2 * Hn * 2);
    __bf16* Wat_b   = (__bf16*)carve((size_t)Hn * Hn * 2);
    __bf16* xbuf    = (__bf16*)carve((size_t)2 * Bn * XKn * 2);      // ping-pong
    float*  hf      = (float*) carve((size_t)Bn * Hn * 4);
    __bf16* hb      = (__bf16*)carve((size_t)Bn * Hn * 2);
    float*  score   = (float*) carve((size_t)Bn * SLOTn * 4);
    float*  ctxf    = (float*) carve((size_t)Bn * Hn * 4);
    float*  h0f     = (float*) carve((size_t)Bn * Hn * 4);
    float*  c1f     = (float*) carve((size_t)Bn * Hn * 4);

    // one-time f32 -> bf16 conversions
    {
        int n;
        n = Bn * Tn * Hn;   cvt_f32_to_bf16<<<(n + 255) / 256, 256, 0, stream>>>(enc, enc_b, n);
        n = G4n * XKn;      cvt_f32_to_bf16<<<(n + 255) / 256, 256, 0, stream>>>(W_ih, Wih_b, n);
        n = SLOTn * 2 * Hn; cvt_f32_to_bf16<<<(n + 255) / 256, 256, 0, stream>>>(W_slot, Wsl_b, n);
        n = Hn * Hn;        cvt_f32_to_bf16<<<(n + 255) / 256, 256, 0, stream>>>(W_attn, Wat_b, n);
    }
    // hoisted attention energies (big WMMA GEMM)
    energies_kernel<<<dim3((Bn * Tn) / 64, Hn / 64), 128, 0, stream>>>(enc_b, Wat_b, b_attn, energ_b);
    // initial carry: embedded=0, ctx0 = enc[b, len-1]
    init_kernel<<<Bn, 256, 0, stream>>>(enc, mask, xbuf);

    for (int t = 0; t < Tn; t++) {
        __bf16* xc = xbuf + (size_t)(t & 1) * Bn * XKn;
        __bf16* xn = xbuf + (size_t)((t + 1) & 1) * Bn * XKn;
        lstm_kernel<<<Hn / 16, 128, 0, stream>>>(xc, Wih_b, b_ih, b_hh, hf, hb);
        slot_gemm  <<<SLOTn / 16, 32, 0, stream>>>(hb, xc, Wsl_b, b_slot, score);
        slot_finish<<<Bn, 256, 0, stream>>>(score, emb, enc_b, out, xn, t);
        attn_kernel<<<Bn, 256, 0, stream>>>(energ_b, hb, enc, mask, ctxf, xn);
        if (t == 0)
            save_h0<<<(Bn * Hn + 255) / 256, 256, 0, stream>>>(hf, ctxf, h0f, c1f);
    }
    intent_kernel<<<Bn, 128, 0, stream>>>(h0f, c1f, W_i1, b_i1, W_i2, b_i2, out);
}
